// MultiHeadSelfAttention_292057776889
// MI455X (gfx1250) — matrix operand order repaired and checked by reading, not compiled, not run on hardware
//
#include <hip/hip_runtime.h>
#include <hip/hip_bf16.h>
#include <stdint.h>

#define N_B   2
#define SEQ   2048
#define HEADS 16
#define HDIM  64
#define EMB   1024
#define SCALE 0.03125f   // 1/sqrt(1024)
#define KTILES (SEQ / 32)

typedef __attribute__((ext_vector_type(16))) __bf16 v16bf;
typedef __attribute__((ext_vector_type(8)))  float  v8f;

union Frag { uint4 q[2]; v16bf v; };

__device__ __forceinline__ v8f wmma_bf16(v16bf a, v16bf b, v8f c) {
  return __builtin_amdgcn_wmma_f32_16x16x32_bf16(false, a, false, b, (short)0, c,
                                                 false, false);
}

// hardware RNE f32->bf16 (selects v_cvt_pk_bf16_f32 when paired)
__device__ __forceinline__ uint16_t f2bf(float x) {
  union { __bf16 h; uint16_t u; } v; v.h = (__bf16)x; return v.u;
}
__device__ __forceinline__ uint32_t pk2(float a, float b) {
  return (uint32_t)f2bf(a) | ((uint32_t)f2bf(b) << 16);
}

// A-fragment: two runs of 8 contiguous f32 (K = g*16.. and g*16+8..), pre-scaled
__device__ __forceinline__ v16bf cvtA_s(const float* __restrict__ r0,
                                        const float* __restrict__ r1, float s) {
  v16bf r;
#pragma unroll
  for (int i = 0; i < 8; ++i) r[i] = (__bf16)(r0[i] * s);
#pragma unroll
  for (int i = 0; i < 8; ++i) r[i + 8] = (__bf16)(r1[i] * s);
  return r;
}

// fragment from two 16B-aligned bf16 runs (LDS or global)
__device__ __forceinline__ v16bf ldfrag(const uint16_t* r0, const uint16_t* r1) {
  Frag f;
  f.q[0] = *(const uint4*)r0;
  f.q[1] = *(const uint4*)r1;
  return f.v;
}

// max across the 16 lanes of a half-wave: VALU-only XOR butterfly via
// v_permlane16_b32 (nibble-select constants encode lane i -> lane i^m).
#if __has_builtin(__builtin_amdgcn_permlane16)
__device__ __forceinline__ float plx(float x, unsigned s0, unsigned s1) {
  union { float f; unsigned u; } a; a.f = x;
  a.u = __builtin_amdgcn_permlane16(a.u, a.u, s0, s1, false, false);
  return a.f;
}
__device__ __forceinline__ float rmax16(float x) {
  x = fmaxf(x, plx(x, 0x67452301u, 0xEFCDAB89u));   // xor 1
  x = fmaxf(x, plx(x, 0x54761032u, 0xDCFE98BAu));   // xor 2
  x = fmaxf(x, plx(x, 0x32107654u, 0xBA98FEDCu));   // xor 4
  x = fmaxf(x, plx(x, 0xFEDCBA98u, 0x76543210u));   // xor 8
  return x;
}
#else
__device__ __forceinline__ float rmax16(float x) {
#pragma unroll
  for (int m = 1; m <= 8; m <<= 1) x = fmaxf(x, __shfl_xor(x, m, 32));
  return x;
}
#endif

// ---------------------------------------------------------------------------
// Kernel 0: one-shot prep. K -> bf16 ([N*S][E]); V -> bf16 transposed per head
// (Vt[((n*H+h)*HDIM + d)][s]); W -> bf16 ([E][E]). Conversion paid once.
// ---------------------------------------------------------------------------
__global__ __launch_bounds__(256) void prep_kernel(
    const float* __restrict__ Kf, const float* __restrict__ Vf,
    const float* __restrict__ Wf, uint16_t* __restrict__ Kb,
    uint16_t* __restrict__ Vt, uint16_t* __restrict__ Wb) {
  const size_t t = (size_t)blockIdx.x * 256 + threadIdx.x;   // elems/4 threads
  const size_t idx = t * 4;

  // K: 4 contiguous elements, coalesced in and out
  {
    const float4 x = *(const float4*)(Kf + idx);
    uint2 o; o.x = pk2(x.x, x.y); o.y = pk2(x.z, x.w);
    *(uint2*)(Kb + idx) = o;
  }
  // Vt: 4 consecutive s at fixed (n,h,d); writes coalesced, reads L2-hot
  {
    const int    s    = (int)(idx % SEQ);
    const size_t rest = idx / SEQ;
    const int    d    = (int)(rest % HDIM);
    const size_t nh   = rest / HDIM;               // n*HEADS + h
    const size_t n = nh / HEADS, h = nh % HEADS;
    const float* src = Vf + (n * SEQ + s) * (size_t)EMB + h * HDIM + d;
    uint2 o;
    o.x = pk2(src[0], src[EMB]);
    o.y = pk2(src[2 * EMB], src[3 * EMB]);
    *(uint2*)(Vt + idx) = o;
  }
  // W: first EMB*EMB/4 threads convert 4 contiguous elements each
  if (t < (size_t)EMB * EMB / 4) {
    const float4 x = *(const float4*)(Wf + idx);
    uint2 o; o.x = pk2(x.x, x.y); o.y = pk2(x.z, x.w);
    *(uint2*)(Wb + idx) = o;
  }
}

// ---------------------------------------------------------------------------
// Kernel 1: flash attention. One wave owns a 16-row q-tile. K fragments are
// software-pipelined one k-tile ahead in registers; V fragments issue at the
// top of the iteration so softmax VALU hides both latencies. P goes through a
// per-wave LDS tile (D-layout -> A-layout); its row sums come from an extra
// WMMA against a ones matrix.
// ---------------------------------------------------------------------------
__global__ __launch_bounds__(256) void flash_attn_kernel(
    const float* __restrict__ Q, const uint16_t* __restrict__ Kb,
    const uint16_t* __restrict__ Vt, uint16_t* __restrict__ attn) {
  __shared__ __align__(16) uint16_t Pb[8][16][32];   // per-wave P tile [q][k]

  const int tid  = threadIdx.x;
  const int wave = tid >> 5, lane = tid & 31;
  const int ln   = lane & 15, g = lane >> 4;
  const int h = blockIdx.y, n = blockIdx.z;
  const int qBase = blockIdx.x * 128 + wave * 16;

  const size_t batOff  = (size_t)n * SEQ;
  const size_t headOff = (size_t)h * HDIM;

  // Q A-fragments, pre-scaled by 1/sqrt(E): rows qBase+ln, d = 0..31 / 32..63
  const float* qrow = Q + (batOff + qBase + ln) * (size_t)EMB + headOff;
  const v16bf QA0 = cvtA_s(qrow + g * 16,      qrow + 8 + g * 16, SCALE);
  const v16bf QA1 = cvtA_s(qrow + 32 + g * 16, qrow + 40 + g * 16, SCALE);

  v16bf ONES;
#pragma unroll
  for (int i = 0; i < 16; ++i) ONES[i] = (__bf16)1.0f;

  v8f acc0 = {0,0,0,0,0,0,0,0}, acc1 = {0,0,0,0,0,0,0,0};
  v8f acc2 = {0,0,0,0,0,0,0,0}, acc3 = {0,0,0,0,0,0,0,0};
  float mrun[8], lrun[8];
#pragma unroll
  for (int r = 0; r < 8; ++r) { mrun[r] = -1e30f; lrun[r] = 0.0f; }

  uint16_t* Pw = &Pb[wave][0][0];
  // per-head V^T base: rows are d (64), columns are s (SEQ)
  const uint16_t* vbase = Vt + ((size_t)n * HEADS + h) * HDIM * SEQ;
  // K row base for subtile c of tile starting at kBase
  const uint16_t* kbase = Kb + batOff * (size_t)EMB + headOff + (size_t)ln * EMB;

  // preamble: K fragments for tile 0 (distinct registers, pipelined)
  v16bf k00, k10, k01, k11;
  {
    const uint16_t* a = kbase;                       // rows kBase+ln
    const uint16_t* b = kbase + (size_t)16 * EMB;    // rows kBase+16+ln
    k00 = ldfrag(a + g * 16, a + 8 + g * 16);
    k10 = ldfrag(a + 32 + g * 16, a + 40 + g * 16);
    k01 = ldfrag(b + g * 16, b + 8 + g * 16);
    k11 = ldfrag(b + 32 + g * 16, b + 40 + g * 16);
  }

  for (int kb = 0; kb < KTILES; ++kb) {
    const int kBase = kb * 32;

    // V fragments first: in flight across QK + softmax
    const uint16_t* v0 = vbase + (size_t)( 0 + ln) * SEQ + kBase + g * 16;
    const uint16_t* v1 = vbase + (size_t)(16 + ln) * SEQ + kBase + g * 16;
    const uint16_t* v2 = vbase + (size_t)(32 + ln) * SEQ + kBase + g * 16;
    const uint16_t* v3 = vbase + (size_t)(48 + ln) * SEQ + kBase + g * 16;
    v16bf vb0 = ldfrag(v0, v0 + 8);
    v16bf vb1 = ldfrag(v1, v1 + 8);
    v16bf vb2 = ldfrag(v2, v2 + 8);
    v16bf vb3 = ldfrag(v3, v3 + 8);

    // energy tiles (scale folded into QA)
    v8f zero = {0,0,0,0,0,0,0,0};
    v8f e0 = wmma_bf16(QA0, k00, zero);
    e0     = wmma_bf16(QA1, k10, e0);
    v8f e1 = wmma_bf16(QA0, k01, zero);
    e1     = wmma_bf16(QA1, k11, e1);

    // pipeline: K fragments for the next tile (wrap keeps loads unconditional)
    {
      const size_t kn = (size_t)(((kb + 1) & (KTILES - 1)) * 32) * EMB;
      const uint16_t* a = kbase + kn;
      const uint16_t* b = a + (size_t)16 * EMB;
      __builtin_prefetch(a + (size_t)32 * EMB, 0, 1);
      k00 = ldfrag(a + g * 16, a + 8 + g * 16);
      k10 = ldfrag(a + 32 + g * 16, a + 40 + g * 16);
      k01 = ldfrag(b + g * 16, b + 8 + g * 16);
      k11 = ldfrag(b + 32 + g * 16, b + 40 + g * 16);
    }

    // online softmax update, write P (bf16) to per-wave LDS tile
#pragma unroll
    for (int r = 0; r < 8; ++r) {
      float rm   = rmax16(fmaxf(e0[r], e1[r]));
      float mnew = fmaxf(mrun[r], rm);
      float al   = __expf(mrun[r] - mnew);
      float p0   = __expf(e0[r] - mnew);
      float p1   = __expf(e1[r] - mnew);
      mrun[r] = mnew;
      lrun[r] *= al;
      acc0[r] *= al; acc1[r] *= al; acc2[r] *= al; acc3[r] *= al;
      const int qr = r + 8 * g;            // this lane-group's row
      Pw[qr * 32 + ln]      = f2bf(p0);
      Pw[qr * 32 + 16 + ln] = f2bf(p1);
    }

    v16bf PA = ldfrag(Pw + ln * 32 + g * 16, Pw + ln * 32 + g * 16 + 8);

    // row sums of P via the matrix pipe: S = P x ones, D-layout matches lrun
    v8f S = wmma_bf16(PA, ONES, zero);
#pragma unroll
    for (int r = 0; r < 8; ++r) lrun[r] += S[r];

    // P (16x32) x V (32x64) with already-resident V fragments
    acc0 = wmma_bf16(PA, vb0, acc0);
    acc1 = wmma_bf16(PA, vb1, acc1);
    acc2 = wmma_bf16(PA, vb2, acc2);
    acc3 = wmma_bf16(PA, vb3, acc3);
  }

  // normalize and emit bf16 attention output
#pragma unroll
  for (int r = 0; r < 8; ++r) {
    float inv = 1.0f / lrun[r];
    const size_t row = (batOff + qBase + r + 8 * g) * (size_t)EMB + headOff + ln;
    attn[row + 0]  = f2bf(acc0[r] * inv);
    attn[row + 16] = f2bf(acc1[r] * inv);
    attn[row + 32] = f2bf(acc2[r] * inv);
    attn[row + 48] = f2bf(acc3[r] * inv);
  }
}

// ---------------------------------------------------------------------------
// Kernel 2: out[m][f] = sum_k attn_bf16[m][k] * Wb[f][k] + b[f]  (torch Linear)
// One wave per 16x16 output tile; pure b128-load + WMMA inner loop.
// ---------------------------------------------------------------------------
__global__ __launch_bounds__(256) void fc_out_kernel(
    const uint16_t* __restrict__ X, const uint16_t* __restrict__ Wb,
    const float* __restrict__ bias, float* __restrict__ out) {
  const int tid  = threadIdx.x;
  const int wave = tid >> 5, lane = tid & 31;
  const int ln   = lane & 15, g = lane >> 4;
  const int fBase = (blockIdx.x * 8 + wave) * 16;
  const int mBase = blockIdx.y * 16;

  v8f c = {0,0,0,0,0,0,0,0};
  const uint16_t* xrow = X  + (size_t)(mBase + ln) * EMB;
  const uint16_t* wrow = Wb + (size_t)(fBase + ln) * EMB + g * 16;

#pragma unroll 8
  for (int kk = 0; kk < EMB / 32; ++kk) {
    v16bf A  = ldfrag(xrow + kk * 32 + g * 16, xrow + kk * 32 + g * 16 + 8);
    v16bf Bf = ldfrag(wrow + kk * 32, wrow + kk * 32 + 8);
    c = wmma_bf16(A, Bf, c);
  }

  float b = bias[fBase + ln];
#pragma unroll
  for (int r = 0; r < 8; ++r)
    out[(size_t)(mBase + r + 8 * g) * EMB + fBase + ln] = c[r] + b;
}

extern "C" void kernel_launch(void* const* d_in, const int* in_sizes, int n_in,
                              void* d_out, int out_size, void* d_ws, size_t ws_size,
                              hipStream_t stream) {
  (void)in_sizes; (void)n_in; (void)out_size; (void)ws_size;
  const float* values = (const float*)d_in[0];
  const float* keys   = (const float*)d_in[1];
  const float* query  = (const float*)d_in[2];
  const float* W      = (const float*)d_in[3];
  const float* bias   = (const float*)d_in[4];

  const size_t elems = (size_t)N_B * SEQ * EMB;     // 4M
  uint16_t* attn = (uint16_t*)d_ws;                 // 8 MB
  uint16_t* Kb   = attn + elems;                    // 8 MB
  uint16_t* Vt   = Kb + elems;                      // 8 MB
  uint16_t* Wb   = Vt + elems;                      // 2 MB

  prep_kernel<<<dim3(elems / 4 / 256), 256, 0, stream>>>(keys, values, W,
                                                         Kb, Vt, Wb);
  flash_attn_kernel<<<dim3(SEQ / 128, HEADS, N_B), 256, 0, stream>>>(
      query, Kb, Vt, attn);
  fc_out_kernel<<<dim3(EMB / 128, (N_B * SEQ) / 16), 256, 0, stream>>>(
      attn, Wb, bias, (float*)d_out);
}
